// MambaLayer_51539607552220
// MI455X (gfx1250) — compile-verified, matrix operand order checked by reading, not run on hardware
//
#include <hip/hip_runtime.h>
#include <hip/hip_bf16.h>

typedef __bf16 v16bf __attribute__((ext_vector_type(16)));
typedef float  v8f   __attribute__((ext_vector_type(8)));
typedef int    v4i   __attribute__((ext_vector_type(4)));

#define BATCH    2
#define SEQ      512
#define D_MODEL  1024
#define D_STATE  64
#define HEADDIM  64
#define D_CONV   4
#define VNUM     3
#define HORIZON  20
#define D_INNER  2048
#define NHEADS   32
#define CONV_DIM 2176
#define D_IN_PROJ 4256
#define D_IN_PROJ_PAD 4352   /* 17 * 256 */
#define ROWS     (BATCH*SEQ)  /* 1024 */
#define RMS_EPS  1e-5f

#if __has_builtin(__builtin_amdgcn_global_load_async_to_lds_b128) && \
    __has_builtin(__builtin_amdgcn_s_wait_asynccnt)
#define HAS_ASYNC_LDS 1
#else
#define HAS_ASYNC_LDS 0
#endif

__device__ __forceinline__ float silu_f(float x) { return x / (1.f + __expf(-x)); }

__device__ __forceinline__ float wave_sum(float v) {
    #pragma unroll
    for (int o = 16; o > 0; o >>= 1) v += __shfl_down(v, o, 32);
    return v;
}

// ---------------------------------------------------------------------------
// f32 -> bf16 convert with row padding: dst[rows_pad, K]; rows >= actual rows
// get zeros so the GEMM never needs bounds checks.
// ---------------------------------------------------------------------------
__global__ __launch_bounds__(256) void cvt_bf16_pad(
    const float* __restrict__ src, __bf16* __restrict__ dst,
    int rows, int K, int rows_pad)
{
    int idx = blockIdx.x * 256 + threadIdx.x;
    int total = rows_pad * K;
    if (idx >= total) return;
    int row = idx / K;
    dst[idx] = (row < rows) ? (__bf16)src[idx] : (__bf16)0.f;
}

// ---------------------------------------------------------------------------
// GEMM: Out[m,n] = sum_k A[m,k] * B[n,k], both bf16 row-major over K.
// Block: 256 threads = 8 waves as 2(M) x 4(N); block tile 32 x 256; K step 32.
// Each wave: 1 A fragment reused by 4 B fragments -> 4 WMMAs / K-step.
// No bounds checks: M % 32 == 0, Npad % 256 == 0, K % 32 == 0 (pre-padded).
// ---------------------------------------------------------------------------
__global__ __launch_bounds__(256) void gemm_bf16_wmma(
    const __bf16* __restrict__ A, const __bf16* __restrict__ B,
    float* __restrict__ Out, int M, int Npad, int K)
{
    __shared__ __bf16 Asm[32 * 32];    // [row][k]
    __shared__ __bf16 Bsm[256 * 32];   // [col][k]

    const int tid  = threadIdx.x;
    const int lane = tid & 31;
    const int wv   = tid >> 5;         // 0..7
    const int mi   = wv >> 2;          // 0..1
    const int ni   = wv & 3;           // 0..3
    const int m0   = blockIdx.x * 32;
    const int n0   = blockIdx.y * 256;
    const int rc   = lane & 15;
    const int khalf = (lane & 16) ? 8 : 0;   // ISA 16-bit fragment K split

    v8f acc0 = {}, acc1 = {}, acc2 = {}, acc3 = {};

    for (int k0 = 0; k0 < K; k0 += 32) {
        // ---- stage A tile 32x32 (1024 bf16): 4 contiguous bf16 per thread
        {
            int idx = tid * 4;                 // 8-byte aligned
            int row = idx >> 5, k = idx & 31;
            *(unsigned long long*)&Asm[idx] =
                *(const unsigned long long*)&A[(size_t)(m0 + row) * K + k0 + k];
        }
        // ---- stage B tile 256x32 (8192 bf16): 4 x b128 per thread
#if HAS_ASYNC_LDS
        #pragma unroll
        for (int i = 0; i < 4; ++i) {
            int chunk = tid + i * 256;         // 1024 chunks of 16B
            int row = chunk >> 2, kq = (chunk & 3) * 8;
            __builtin_amdgcn_global_load_async_to_lds_b128(
                (v4i*)&B[(size_t)(n0 + row) * K + k0 + kq],
                (__attribute__((address_space(3))) v4i*)&Bsm[row * 32 + kq],
                0, 0);
        }
        __builtin_amdgcn_s_wait_asynccnt(0);
#else
        #pragma unroll
        for (int i = 0; i < 4; ++i) {
            int chunk = tid + i * 256;
            int row = chunk >> 2, kq = (chunk & 3) * 8;
            *(ulonglong2*)&Bsm[row * 32 + kq] =
                *(const ulonglong2*)&B[(size_t)(n0 + row) * K + k0 + kq];
        }
#endif
        __syncthreads();

        v16bf af, bf0, bf1, bf2, bf3;
        #pragma unroll
        for (int e = 0; e < 16; ++e) {
            int k = ((e >> 3) << 4) + khalf + (e & 7);
            af[e]  = Asm[(mi * 16 + rc) * 32 + k];
            bf0[e] = Bsm[(ni * 64 +  0 + rc) * 32 + k];
            bf1[e] = Bsm[(ni * 64 + 16 + rc) * 32 + k];
            bf2[e] = Bsm[(ni * 64 + 32 + rc) * 32 + k];
            bf3[e] = Bsm[(ni * 64 + 48 + rc) * 32 + k];
        }
        acc0 = __builtin_amdgcn_wmma_f32_16x16x32_bf16(false, af, false, bf0, (short)0, acc0, false, false);
        acc1 = __builtin_amdgcn_wmma_f32_16x16x32_bf16(false, af, false, bf1, (short)0, acc1, false, false);
        acc2 = __builtin_amdgcn_wmma_f32_16x16x32_bf16(false, af, false, bf2, (short)0, acc2, false, false);
        acc3 = __builtin_amdgcn_wmma_f32_16x16x32_bf16(false, af, false, bf3, (short)0, acc3, false, false);
        __syncthreads();
    }

    // store: VGPR r -> M = r (+8 for lanes 16-31), N = lane&15
    const int mrow = m0 + mi * 16 + ((lane & 16) ? 8 : 0);
    const int nb   = n0 + ni * 64 + rc;
    #pragma unroll
    for (int r = 0; r < 8; ++r) {
        Out[(size_t)(mrow + r) * Npad + nb +  0] = acc0[r];
        Out[(size_t)(mrow + r) * Npad + nb + 16] = acc1[r];
        Out[(size_t)(mrow + r) * Npad + nb + 32] = acc2[r];
        Out[(size_t)(mrow + r) * Npad + nb + 48] = acc3[r];
    }
}

// ---------------------------------------------------------------------------
// Depthwise causal conv(4) + bias + SiLU; softplus(dt + dt_bias).
// zxbcdt has padded stride D_IN_PROJ_PAD. One block per (b,s) row.
// ---------------------------------------------------------------------------
__global__ __launch_bounds__(256) void conv_silu_dt(
    const float* __restrict__ zxbcdt, const float* __restrict__ conv_w,
    const float* __restrict__ conv_b, const float* __restrict__ dt_bias,
    float* __restrict__ xBC, float* __restrict__ dt_soft)
{
    const int row = blockIdx.x;           // b*SEQ + s
    const int b = row / SEQ, s = row % SEQ;
    for (int c = threadIdx.x; c < CONV_DIM; c += 256) {
        float acc = conv_b[c];
        #pragma unroll
        for (int k = 0; k < D_CONV; ++k) {
            int ss = s + k - (D_CONV - 1);
            if (ss >= 0)
                acc += conv_w[c * D_CONV + k] *
                       zxbcdt[(size_t)(b * SEQ + ss) * D_IN_PROJ_PAD + D_INNER + c];
        }
        xBC[(size_t)row * CONV_DIM + c] = silu_f(acc);
    }
    if (threadIdx.x < NHEADS) {
        int h = threadIdx.x;
        float v = zxbcdt[(size_t)row * D_IN_PROJ_PAD + D_INNER + CONV_DIM + h] + dt_bias[h];
        dt_soft[row * NHEADS + h] = (v > 20.f) ? v : log1pf(__expf(v));
    }
}

// ---------------------------------------------------------------------------
// Sequential SSM scan. One block per (b, head); state in registers.
// ---------------------------------------------------------------------------
__global__ __launch_bounds__(256) void ssm_scan(
    const float* __restrict__ xBC, const float* __restrict__ dt_soft,
    const float* __restrict__ A_log, const float* __restrict__ Dvec,
    float* __restrict__ y_raw)
{
    const int bh = blockIdx.x;            // b*NHEADS + h
    const int b = bh / NHEADS, h = bh % NHEADS;
    const int tid = threadIdx.x;
    const int p   = tid & 63;
    const int ng  = tid >> 6;             // 0..3
    const int nb  = ng * 16;

    __shared__ float Bsh[D_STATE], Csh[D_STATE], red[4 * HEADDIM];

    const float Ah = -__expf(A_log[h]);
    const float Dh = Dvec[h];
    float hs[16];
    #pragma unroll
    for (int j = 0; j < 16; ++j) hs[j] = 0.f;

    for (int s = 0; s < SEQ; ++s) {
        const size_t row = (size_t)(b * SEQ + s);
        if (tid < D_STATE)
            Bsh[tid] = xBC[row * CONV_DIM + D_INNER + tid];
        else if (tid < 2 * D_STATE)
            Csh[tid - D_STATE] = xBC[row * CONV_DIM + D_INNER + D_STATE + (tid - D_STATE)];
        __syncthreads();

        const float dt = dt_soft[row * NHEADS + h];
        const float dA = __expf(dt * Ah);
        const float xp = xBC[row * CONV_DIM + h * HEADDIM + p];
        const float sc = dt * xp;

        float acc = 0.f;
        #pragma unroll
        for (int j = 0; j < 16; ++j) {
            const int n = nb + j;
            hs[j] = hs[j] * dA + sc * Bsh[n];
            acc += hs[j] * Csh[n];
        }
        red[ng * HEADDIM + p] = acc;
        __syncthreads();
        if (ng == 0) {
            float y = red[p] + red[HEADDIM + p] + red[2 * HEADDIM + p] +
                      red[3 * HEADDIM + p] + Dh * xp;
            y_raw[row * D_INNER + h * HEADDIM + p] = y;
        }
        __syncthreads();
    }
}

// ---------------------------------------------------------------------------
// y = y_raw * silu(z); RMSNorm; * norm_w -> bf16 (feeds out_proj WMMA GEMM).
// ---------------------------------------------------------------------------
__global__ __launch_bounds__(256) void gate_rmsnorm(
    const float* __restrict__ y_raw, const float* __restrict__ zxbcdt,
    const float* __restrict__ norm_w, __bf16* __restrict__ yn)
{
    const int row = blockIdx.x;
    __shared__ float rsum[256];
    float vals[8];
    float ss = 0.f;
    #pragma unroll
    for (int i = 0; i < 8; ++i) {
        int d = threadIdx.x + i * 256;
        float z = zxbcdt[(size_t)row * D_IN_PROJ_PAD + d];
        float v = y_raw[(size_t)row * D_INNER + d] * silu_f(z);
        vals[i] = v; ss += v * v;
    }
    rsum[threadIdx.x] = ss; __syncthreads();
    for (int o = 128; o > 0; o >>= 1) {
        if (threadIdx.x < o) rsum[threadIdx.x] += rsum[threadIdx.x + o];
        __syncthreads();
    }
    const float r = rsqrtf(rsum[0] / (float)D_INNER + RMS_EPS);
    #pragma unroll
    for (int i = 0; i < 8; ++i) {
        int d = threadIdx.x + i * 256;
        yn[(size_t)row * D_INNER + d] = (__bf16)(vals[i] * r * norm_w[d]);
    }
}

// ---------------------------------------------------------------------------
// Decoders: wave-per-row GEMV, lane-strided + shfl reduction (wave32).
// ---------------------------------------------------------------------------
__global__ __launch_bounds__(256) void decode_current(
    const float* __restrict__ hidden, const float* __restrict__ wc,
    const float* __restrict__ bc, float* __restrict__ outc)
{
    const int wv = threadIdx.x >> 5, lane = threadIdx.x & 31;
    const int m = blockIdx.x * 8 + wv;          // ROWS waves total
    float a0 = 0.f, a1 = 0.f, a2 = 0.f;
    for (int d = lane; d < D_MODEL; d += 32) {
        float hv = hidden[(size_t)m * D_MODEL + d];
        a0 += hv * wc[0 * D_MODEL + d];
        a1 += hv * wc[1 * D_MODEL + d];
        a2 += hv * wc[2 * D_MODEL + d];
    }
    a0 = wave_sum(a0); a1 = wave_sum(a1); a2 = wave_sum(a2);
    if (lane == 0) {
        outc[m * VNUM + 0] = a0 + bc[0];
        outc[m * VNUM + 1] = a1 + bc[1];
        outc[m * VNUM + 2] = a2 + bc[2];
    }
}

__global__ __launch_bounds__(256) void decode_future(
    const float* __restrict__ hidden, const float* __restrict__ wf,
    const float* __restrict__ bf, float* __restrict__ outf)
{
    const int wv = threadIdx.x >> 5, lane = threadIdx.x & 31;
    const int o = blockIdx.x * 8 + wv;          // 120 outputs = 120 waves
    if (o >= BATCH * HORIZON * VNUM) return;
    const int b = o / (HORIZON * VNUM), hv = o % (HORIZON * VNUM);
    const size_t row = (size_t)(b * SEQ + SEQ - 1);
    float a = 0.f;
    for (int d = lane; d < D_MODEL; d += 32)
        a += hidden[row * D_MODEL + d] * wf[(size_t)hv * D_MODEL + d];
    a = wave_sum(a);
    if (lane == 0) outf[o] = a + bf[hv];
}

// ---------------------------------------------------------------------------
extern "C" void kernel_launch(void* const* d_in, const int* in_sizes, int n_in,
                              void* d_out, int out_size, void* d_ws, size_t ws_size,
                              hipStream_t stream)
{
    const float* embed     = (const float*)d_in[0];
    const float* in_proj_w = (const float*)d_in[1];
    const float* conv_w    = (const float*)d_in[2];
    const float* conv_b    = (const float*)d_in[3];
    const float* dt_bias   = (const float*)d_in[4];
    const float* A_log     = (const float*)d_in[5];
    const float* Dvec      = (const float*)d_in[6];
    const float* norm_w    = (const float*)d_in[7];
    const float* out_proj_w= (const float*)d_in[8];
    const float* dec_cur_w = (const float*)d_in[9];
    const float* dec_cur_b = (const float*)d_in[10];
    const float* dec_fut_w = (const float*)d_in[11];
    const float* dec_fut_b = (const float*)d_in[12];
    float* out = (float*)d_out;

    // ---- workspace layout
    char* p = (char*)d_ws;
    float* zxbcdt  = (float*)p;  p += (size_t)ROWS * D_IN_PROJ_PAD * 4;
    float* xBC     = (float*)p;  p += (size_t)ROWS * CONV_DIM * 4;
    float* dt_soft = (float*)p;  p += (size_t)ROWS * NHEADS * 4;
    float* y_raw   = (float*)p;  p += (size_t)ROWS * D_INNER * 4;
    float* hidden  = (float*)p;  p += (size_t)ROWS * D_MODEL * 4;
    __bf16* embed_h = (__bf16*)p; p += (size_t)ROWS * D_MODEL * 2;
    __bf16* inw_h   = (__bf16*)p; p += (size_t)D_IN_PROJ_PAD * D_MODEL * 2;
    __bf16* outw_h  = (__bf16*)p; p += (size_t)D_MODEL * D_INNER * 2;
    __bf16* yn_h    = (__bf16*)p; p += (size_t)ROWS * D_INNER * 2;

    // 0) weight / activation bf16 conversion (padded rows -> zero)
    {
        int n;
        n = ROWS * D_MODEL;
        cvt_bf16_pad<<<(n + 255) / 256, 256, 0, stream>>>(embed, embed_h,
            ROWS, D_MODEL, ROWS);
        n = D_IN_PROJ_PAD * D_MODEL;
        cvt_bf16_pad<<<(n + 255) / 256, 256, 0, stream>>>(in_proj_w, inw_h,
            D_IN_PROJ, D_MODEL, D_IN_PROJ_PAD);
        n = D_MODEL * D_INNER;
        cvt_bf16_pad<<<(n + 255) / 256, 256, 0, stream>>>(out_proj_w, outw_h,
            D_MODEL, D_INNER, D_MODEL);
    }

    // 1) in_proj: zxbcdt[1024, 4352pad] = embed @ in_proj_w^T
    gemm_bf16_wmma<<<dim3(ROWS / 32, D_IN_PROJ_PAD / 256), 256, 0, stream>>>(
        embed_h, inw_h, zxbcdt, ROWS, D_IN_PROJ_PAD, D_MODEL);

    // 2) conv + silu + softplus(dt)
    conv_silu_dt<<<ROWS, 256, 0, stream>>>(zxbcdt, conv_w, conv_b, dt_bias,
                                           xBC, dt_soft);

    // 3) sequential SSM scan (+ D*x skip)
    ssm_scan<<<BATCH * NHEADS, 256, 0, stream>>>(xBC, dt_soft, A_log, Dvec, y_raw);

    // 4) gate * silu(z), RMSNorm -> bf16
    gate_rmsnorm<<<ROWS, 256, 0, stream>>>(y_raw, zxbcdt, norm_w, yn_h);

    // 5) out_proj: hidden[1024,1024] = yn @ out_proj_w^T
    gemm_bf16_wmma<<<dim3(ROWS / 32, D_MODEL / 256), 256, 0, stream>>>(
        yn_h, outw_h, hidden, ROWS, D_MODEL, D_INNER);

    // 6) decoders -> d_out (pred_current [2,512,3] then pred_future [2,20,3])
    decode_current<<<ROWS / 8, 256, 0, stream>>>(hidden, dec_cur_w, dec_cur_b, out);
    decode_future<<<(BATCH * HORIZON * VNUM + 7) / 8, 256, 0, stream>>>(
        hidden, dec_fut_w, dec_fut_b, out + ROWS * VNUM);
}
